// AdditiveCoupling_87076166960207
// MI455X (gfx1250) — matrix operand order repaired and checked by reading, not compiled, not run on hardware
//
#include <hip/hip_runtime.h>
#include <hip/hip_bf16.h>

// ---------------------------------------------------------------------------
// Additive coupling layer, fused for MI455X (gfx1250, wave32, WMMA).
//   B=32768 rows, D=2048, half=1024, MID=128.
//   HBM floor: read x (256MB) + write y (256MB) = 512MB -> ~22us @ 23.3TB/s.
//   bf16 WMMA (16x16x32, f32 accum) keeps the 18.3 GFLOP MLP far under that.
// ---------------------------------------------------------------------------

#define B_ROWS 32768
#define DIM    2048
#define HALF   1024
#define MID    128
#define TM     128   // rows per workgroup (8 waves x 16-row WMMA slab)
#define KC     64    // K-chunk for GEMM1 staging

typedef __attribute__((ext_vector_type(16))) __bf16 v16bf;
typedef __attribute__((ext_vector_type(8)))  __bf16 v8bf;
typedef __attribute__((ext_vector_type(8)))  float  v8f;

// ---- CDNA5 WMMA fragment loaders (ISA 7.12.2 layouts, wave32) -------------
// A (16x32 bf16): lane l: M = l&15; K = {8*(l>>4)..+7} and {16+8*(l>>4)..+7}
//   -> two contiguous 16B LDS loads (ds_load_b128).
__device__ __forceinline__ v16bf load_frag_A(const __bf16* base, int m0,
                                             int ldk, int kb, int lane) {
  const int lo = lane & 15, hi = lane >> 4;
  const __bf16* p = base + (m0 + lo) * ldk + kb + 8 * hi;
  union { v16bf v; v8bf h[2]; } u;
  u.h[0] = *(const v8bf*)p;
  u.h[1] = *(const v8bf*)(p + 16);
  return u.v;
}

// B (32x16 bf16, K-major in LDS): lane l holds column N = l&15,
// K = {8*(l>>4)..+7} and {16+8*(l>>4)..+7} (same order as A) -> strided LDS loads.
__device__ __forceinline__ v16bf load_frag_B(const __bf16* base, int kb,
                                             int ldn, int nb, int lane) {
  const int lo = lane & 15, hi = lane >> 4;
  const __bf16* p = base + (kb + 8 * hi) * ldn + nb + lo;
  v16bf v;
  for (int s = 0; s < 8; ++s) {
    v[s] = p[s * ldn];
    v[s + 8] = p[(s + 16) * ldn];
  }
  return v;
}

// ---- prep: weights -> bf16 K-major transposes; log_det_J passthrough ------
__global__ void prep_kernel(const float* __restrict__ W0,
                            const float* __restrict__ W1,
                            const float* __restrict__ W2,
                            const float* __restrict__ ldj,
                            __bf16* __restrict__ W0T,   // [1024][128]
                            __bf16* __restrict__ W1T,   // [128][128]
                            __bf16* __restrict__ W2T,   // [128][1024]
                            float* __restrict__ out_ldj) {
  const int i = blockIdx.x * 256 + threadIdx.x;
  if (i < HALF * MID) {               // W0 is [128][1024] -> W0T[k][n]
    const int k = i >> 7, n = i & 127;
    W0T[i] = (__bf16)W0[n * HALF + k];
  }
  if (i < MID * MID) {                // W1 is [128][128]  -> W1T[k][n]
    const int k = i >> 7, n = i & 127;
    W1T[i] = (__bf16)W1[n * MID + k];
  }
  if (i < MID * HALF) {               // W2 is [1024][128] -> W2T[k][n]
    const int k = i >> 10, n = i & 1023;
    W2T[i] = (__bf16)W2[n * MID + k];
  }
  if (i == 0) out_ldj[0] = ldj[0];
}

// ---- fused coupling kernel ------------------------------------------------
__global__ __launch_bounds__(256) void coupling_kernel(
    const float* __restrict__ x,
    const float* __restrict__ b0, const float* __restrict__ b1,
    const float* __restrict__ b2,
    const int* __restrict__ perm_bank, const int* __restrict__ arm_idx,
    const __bf16* __restrict__ W0T, const __bf16* __restrict__ W1T,
    const __bf16* __restrict__ W2T,
    float* __restrict__ y) {
  __shared__ __align__(16) __bf16 Abuf[TM * KC];     // 16 KB
  __shared__ __align__(16) __bf16 Bbuf[MID * MID];   // 32 KB (B tiles)
  __shared__ __align__(16) __bf16 Hbuf[TM * MID];    // 32 KB (h / h2)
  __shared__ int   permS[DIM];                       // 8 KB
  __shared__ float b0S[MID], b1S[MID], b2S[HALF];    // 5 KB

  const int tid  = threadIdx.x;
  const int wid  = tid >> 5;          // wave id 0..7: owns rows [16w,16w+16)
  const int lane = tid & 31;
  const int lo   = lane & 15, hi = lane >> 4;
  const int row0 = blockIdx.x * TM;

  { // stage permutation + biases once
    const int arm = arm_idx[0];
    const int* perm = perm_bank + arm * DIM;
    for (int i = tid; i < DIM; i += 256) permS[i] = perm[i];
    for (int i = tid; i < MID; i += 256) { b0S[i] = b0[i]; b1S[i] = b1[i]; }
    for (int i = tid; i < HALF; i += 256) b2S[i] = b2[i];
  }
  __syncthreads();

  const v8f vzero = {0.f, 0.f, 0.f, 0.f, 0.f, 0.f, 0.f, 0.f};
  v8f acc[8];

  // ---------------- GEMM1: h = relu(x1 @ W0^T + b0) ------------------------
  for (int nt = 0; nt < 8; ++nt) acc[nt] = vzero;
  for (int kc = 0; kc < HALF; kc += KC) {
    // stage A chunk: gathered, deinterleaved x1 -> bf16 LDS [m][k]
    for (int i = tid; i < TM * KC; i += 256) {
      const int m = i >> 6, k = i & (KC - 1);
      const int col = permS[2 * (kc + k)];                 // even -> x1
      Abuf[m * KC + k] = (__bf16)x[(size_t)(row0 + m) * DIM + col];
    }
    // stage B chunk: W0T rows kc..kc+KC (linear 16B copies)
    for (int i = tid; i < (KC * MID) / 8; i += 256)
      *(v8bf*)&Bbuf[i * 8] = *(const v8bf*)&W0T[(size_t)kc * MID + i * 8];
    __syncthreads();
    for (int ks = 0; ks < KC; ks += 32) {
      const v16bf a = load_frag_A(Abuf, wid * 16, KC, ks, lane);
      for (int nt = 0; nt < 8; ++nt) {
        const v16bf bf = load_frag_B(Bbuf, ks, MID, nt * 16, lane);
        acc[nt] = __builtin_amdgcn_wmma_f32_16x16x32_bf16(
            false, a, false, bf, (short)0, acc[nt], false, false);
      }
    }
    __syncthreads();
  }
  // bias + relu -> Hbuf (C-layout: lane->N=16*nt+lo, vgpr r->M=8*hi+r)
  for (int nt = 0; nt < 8; ++nt) {
    const float bias = b0S[nt * 16 + lo];
    for (int r = 0; r < 8; ++r) {
      float v = acc[nt][r] + bias;
      v = v > 0.f ? v : 0.f;
      Hbuf[(wid * 16 + hi * 8 + r) * MID + nt * 16 + lo] = (__bf16)v;
    }
  }
  __syncthreads();

  // ---------------- GEMM2: h2 = relu(h @ W1^T + b1) ------------------------
  for (int i = tid; i < (MID * MID) / 8; i += 256)
    *(v8bf*)&Bbuf[i * 8] = *(const v8bf*)&W1T[i * 8];
  for (int nt = 0; nt < 8; ++nt) acc[nt] = vzero;
  __syncthreads();
  for (int ks = 0; ks < MID; ks += 32) {
    const v16bf a = load_frag_A(Hbuf, wid * 16, MID, ks, lane);
    for (int nt = 0; nt < 8; ++nt) {
      const v16bf bf = load_frag_B(Bbuf, ks, MID, nt * 16, lane);
      acc[nt] = __builtin_amdgcn_wmma_f32_16x16x32_bf16(
          false, a, false, bf, (short)0, acc[nt], false, false);
    }
  }
  __syncthreads();   // all reads of Hbuf/Bbuf done before overwrite
  for (int nt = 0; nt < 8; ++nt) {
    const float bias = b1S[nt * 16 + lo];
    for (int r = 0; r < 8; ++r) {
      float v = acc[nt][r] + bias;
      v = v > 0.f ? v : 0.f;
      Hbuf[(wid * 16 + hi * 8 + r) * MID + nt * 16 + lo] = (__bf16)v;
    }
  }
  __syncthreads();

  // ------- GEMM3: m = h2 @ W2^T + b2, fused scatter epilogue ---------------
  for (int nb = 0; nb < HALF; nb += 128) {
    for (int i = tid; i < (MID * 128) / 8; i += 256) {
      const int k = i >> 4, n = (i * 8) & 127;
      *(v8bf*)&Bbuf[k * 128 + n] =
          *(const v8bf*)&W2T[(size_t)k * HALF + nb + n];
    }
    __syncthreads();
    for (int nt = 0; nt < 8; ++nt) acc[nt] = vzero;
    for (int ks = 0; ks < MID; ks += 32) {
      const v16bf a = load_frag_A(Hbuf, wid * 16, MID, ks, lane);
      for (int nt = 0; nt < 8; ++nt) {
        const v16bf bf = load_frag_B(Bbuf, ks, 128, nt * 16, lane);
        acc[nt] = __builtin_amdgcn_wmma_f32_16x16x32_bf16(
            false, a, false, bf, (short)0, acc[nt], false, false);
      }
    }
    // epilogue straight from accumulators:
    //   y[:, perm[2j]]   = x1[:, j]          (passthrough)
    //   y[:, perm[2j+1]] = x2[:, j] + m[:, j]
    // x rows are L2-resident from the GEMM1 gather -> cheap re-read.
    for (int nt = 0; nt < 8; ++nt) {
      const int j    = nb + nt * 16 + lo;
      const int colA = permS[2 * j];
      const int colB = permS[2 * j + 1];
      const float bias = b2S[j];
      for (int r = 0; r < 8; ++r) {
        const int m = wid * 16 + hi * 8 + r;
        const size_t roff = (size_t)(row0 + m) * DIM;
        const float x1v = x[roff + colA];
        const float x2v = x[roff + colB];
        y[roff + colA] = x1v;
        y[roff + colB] = x2v + acc[nt][r] + bias;
      }
    }
    __syncthreads();
  }
}

// ---------------------------------------------------------------------------
extern "C" void kernel_launch(void* const* d_in, const int* in_sizes, int n_in,
                              void* d_out, int out_size, void* d_ws,
                              size_t ws_size, hipStream_t stream) {
  const float* x   = (const float*)d_in[0];
  const float* ldj = (const float*)d_in[1];
  const float* W0  = (const float*)d_in[2];
  const float* b0  = (const float*)d_in[3];
  const float* W1  = (const float*)d_in[4];
  const float* b1  = (const float*)d_in[5];
  const float* W2  = (const float*)d_in[6];
  const float* b2  = (const float*)d_in[7];
  const int* perm_bank = (const int*)d_in[8];
  // d_in[9] = inv_perm_bank (unused: scatter via perm is its inverse)
  const int* arm = (const int*)d_in[10];
  float* y = (float*)d_out;

  __bf16* W0T = (__bf16*)d_ws;                 // 1024*128
  __bf16* W1T = W0T + HALF * MID;              // 128*128
  __bf16* W2T = W1T + MID * MID;               // 128*1024

  prep_kernel<<<(HALF * MID + 255) / 256, 256, 0, stream>>>(
      W0, W1, W2, ldj, W0T, W1T, W2T, y + (size_t)B_ROWS * DIM);

  coupling_kernel<<<B_ROWS / TM, 256, 0, stream>>>(
      x, b0, b1, b2, perm_bank, arm, W0T, W1T, W2T, y);
}
